// ImageToTextConvProj_16836271800796
// MI455X (gfx1250) — matrix operand order repaired and checked by reading, not compiled, not run on hardware
//
#include <hip/hip_runtime.h>
#include <math.h>

// ---------------------------------------------------------------------------
// ImageToTextConvProj on gfx1250 (MI455X, wave32, WMMA)
//
// Bi=32 images, Bc=128 captions, T=60, LAT=1024, D=300, H=128, O=128.
// Dominant work: per-image dynamic conv = GEMM  M=128(O) x N=7936(Bc*62)
// x K=320*k (D padded 300->320 per tap), f16 operands, f32 accumulate via
// v_wmma_f32_16x16x32_f16.  Conv bias cancels exactly under BN over (Bc,T)
// and max_t relu(BN(x)) is monotone per channel, so the epilogue is fused to
// per-channel {sum,sumsq} + per-(b,o) {max,min}; C is never materialized.
//
// Round-3: fix async-to-LDS builtin signature — it takes
// (int4 addrspace(1)* src, int4 addrspace(3)* dst, imm offset, imm cpol).
// ---------------------------------------------------------------------------

typedef __attribute__((ext_vector_type(16))) _Float16 v16h;
typedef __attribute__((ext_vector_type(8)))  _Float16 v8h;
typedef __attribute__((ext_vector_type(8)))  float    v8f;
typedef __attribute__((ext_vector_type(4)))  int      v4i;

#define BI   32
#define BC   128
#define TT   60
#define TOUT 62
#define XPAD 64      // padded time axis (pad=2 both sides)
#define DD   300
#define DPAD 320     // D padded to multiple of 32
#define HH   128
#define OO   128
#define NCOL (BC*TOUT)        // 7936 = 496 * 16 exactly
#define NTILES 496
#define EPS  1e-5f

// per-image kern sections (halves): k1 128*320, k2 128*640, k3 128*960
#define KERN_PER_IMG (OO*(DPAD*1 + DPAD*2 + DPAD*3))   // 128*1920 = 245760

#if defined(__has_builtin)
#  if __has_builtin(__builtin_amdgcn_global_load_async_to_lds_b128)
#    define HAVE_ASYNC_LDS 1
#  else
#    define HAVE_ASYNC_LDS 0
#  endif
#  if __has_builtin(__builtin_amdgcn_s_wait_asynccnt)
#    define HAVE_WAIT_ASYNC 1
#  else
#    define HAVE_WAIT_ASYNC 0
#  endif
#else
#  define HAVE_ASYNC_LDS 0
#  define HAVE_WAIT_ASYNC 0
#endif

__device__ __forceinline__ void wait_async0() {
#if HAVE_ASYNC_LDS
#  if HAVE_WAIT_ASYNC
  __builtin_amdgcn_s_wait_asynccnt(0);
#  else
  asm volatile("s_wait_asynccnt 0" ::: "memory");
#  endif
#endif
}

#if HAVE_ASYNC_LDS
typedef __attribute__((address_space(1))) v4i* gv4i_p;
typedef __attribute__((address_space(3))) v4i* lv4i_p;
// 16-byte async copy global -> LDS (ASYNCcnt-tracked).
__device__ __forceinline__ void async_cp16(const _Float16* g, _Float16* l) {
  __builtin_amdgcn_global_load_async_to_lds_b128((gv4i_p)g, (lv4i_p)l, 0, 0);
}
#endif

__device__ __forceinline__ void atomicMaxF(float* a, float v) {
  if (v >= 0.f) atomicMax((int*)a, __float_as_int(v));
  else          atomicMin((unsigned int*)a, __float_as_uint(v));
}
__device__ __forceinline__ void atomicMinF(float* a, float v) {
  if (v >= 0.f) atomicMin((int*)a, __float_as_int(v));
  else          atomicMax((unsigned int*)a, __float_as_uint(v));
}

// ---------------------------------------------------------------------------
// K0: workspace init (harness does not re-poison between replays)
// ---------------------------------------------------------------------------
__global__ void __launch_bounds__(256)
k_init(float* gsum, float* gsq, float* gmax, float* gmin) {
  long idx = (long)blockIdx.x * 256 + threadIdx.x;
  const float INF = __int_as_float(0x7f800000);
  if (idx < (long)BI*3*OO*BC) { gmax[idx] = -INF; gmin[idx] = INF; }
  if (idx < BI*3*OO)          { gsum[idx] = 0.f;  gsq[idx]  = 0.f; }
}

// ---------------------------------------------------------------------------
// K1: ic[i][h] = mean_r(img[i][r][:]) @ adapt_w + adapt_b   (one block/image)
// ---------------------------------------------------------------------------
__global__ void __launch_bounds__(256)
k_adapt(const float* __restrict__ img, const float* __restrict__ aw,
        const float* __restrict__ ab, float* __restrict__ ic) {
  __shared__ float m[1024];
  int i = blockIdx.x, tid = threadIdx.x;
  for (int l = tid; l < 1024; l += 256) {
    float s = 0.f;
    const float* p = img + (long)i*36*1024 + l;
    #pragma unroll 4
    for (int r = 0; r < 36; ++r) s += p[r*1024];
    m[l] = s * (1.f/36.f);
  }
  __syncthreads();
  if (tid < HH) {
    float s = ab[tid];
    for (int l = 0; l < 1024; ++l) s += m[l] * aw[l*HH + tid];
    ic[i*HH + tid] = s;
  }
}

// ---------------------------------------------------------------------------
// K2: capT[b][x][c] (f16), x in [0,64) padded time, c in [0,320) padded chan
// ---------------------------------------------------------------------------
__global__ void __launch_bounds__(256)
k_capT(const float* __restrict__ cap, _Float16* __restrict__ capT) {
  long idx = (long)blockIdx.x * 256 + threadIdx.x;      // 128*64*320 total
  if (idx >= (long)BC*XPAD*DPAD) return;
  int c = (int)(idx % DPAD);
  long r = idx / DPAD;
  int x = (int)(r % XPAD);
  int b = (int)(r / XPAD);
  float v = 0.f;
  if (c < DD && x >= 2 && x < 2 + TT)
    v = cap[((long)b*TT + (x - 2))*DD + c];
  capT[idx] = (_Float16)v;
}

// ---------------------------------------------------------------------------
// K3: hypernet kernel generation + kernel-BN -> f16 kernA
//     grid (o=128, kidx=3, img=32); kern raw value n = o*D*k + c*k + j
//     stored as kernA[img][sec(kidx)][o][ j*320 + c ], zero pad c in [300,320)
// ---------------------------------------------------------------------------
struct GenP {
  const float* ic;
  const float* kw[3]; const float* kb[3];
  const float* g[3];  const float* be[3];
  _Float16* kernA;
};

__global__ void __launch_bounds__(256)
k_gen(GenP p) {
  int o = blockIdx.x, kidx = blockIdx.y, img = blockIdx.z;
  int k = kidx + 1, DK = DD * k;
  const float* kw = p.kw[kidx]; const float* kb = p.kb[kidx];
  __shared__ float icsh[HH];
  __shared__ float buf[DD*3];
  __shared__ float rs[256], rq[256];
  int tid = threadIdx.x;
  if (tid < HH) icsh[tid] = p.ic[img*HH + tid];
  __syncthreads();

  long Ntot = (long)OO * DK;           // column stride of kw rows
  float s = 0.f, q = 0.f;
  for (int idx = tid; idx < DK; idx += 256) {
    long n = (long)o * DK + idx;
    float v = kb[n];
    const float* col = kw + n;
    for (int h = 0; h < HH; ++h) v += icsh[h] * col[(long)h * Ntot];
    buf[idx] = v; s += v; q += v * v;
  }
  rs[tid] = s; rq[tid] = q; __syncthreads();
  for (int off = 128; off > 0; off >>= 1) {
    if (tid < off) { rs[tid] += rs[tid+off]; rq[tid] += rq[tid+off]; }
    __syncthreads();
  }
  float inv = 1.f / (float)DK;
  float mean = rs[0] * inv;
  float var  = rq[0] * inv - mean*mean;
  float sc = rsqrtf(var + EPS) * p.g[kidx][o];
  float sh = p.be[kidx][o] - mean * sc;

  long sec = (kidx == 0) ? 0 : (kidx == 1 ? (long)OO*DPAD : (long)OO*DPAD*3);
  long base = (long)img * KERN_PER_IMG + sec + (long)o * (DPAD*k);
  for (int idx = tid; idx < DK; idx += 256) {
    int c = idx / k, j = idx - c*k;
    p.kernA[base + j*DPAD + c] = (_Float16)(buf[idx]*sc + sh);
  }
  for (int z = tid; z < (DPAD-DD)*k; z += 256) {   // zero the K pad
    int c = DD + z / k, j = z - (z/k)*k;
    p.kernA[base + j*DPAD + c] = (_Float16)0.f;
  }
}

// ---------------------------------------------------------------------------
// K4: WMMA conv + fused reductions.
//     grid (62 n-groups, kidx=3, img=32), 256 threads = 8 waves.
//     wave w owns n-tile = bx*8+w (16 columns), all 8 M-tiles (O=128).
//     LDS A-slice double-buffered: one barrier per K-step.
// ---------------------------------------------------------------------------
struct ConvP {
  const _Float16* kernA;
  const _Float16* capT;
  float* gsum; float* gsq; float* gmax; float* gmin;
};

__global__ void __launch_bounds__(256)
k_conv(ConvP p) {
  __shared__ _Float16 Asl[2][OO*32];   // 2 x (128 rows x 32 halves) = 16KB
  __shared__ float ssum[OO], ssq[OO];

  int tid  = threadIdx.x;
  int lane = tid & 31, w = tid >> 5;
  int kidx = blockIdx.y, img = blockIdx.z;
  int k = kidx + 1, Kt = DPAD * k, ksteps = Kt / 32;
  int delta = (kidx == 0) ? 1 : 0;
  long sec = (kidx == 0) ? 0 : (kidx == 1 ? (long)OO*DPAD : (long)OO*DPAD*3);
  const _Float16* Abase = p.kernA + (long)img * KERN_PER_IMG + sec;

  if (tid < OO) { ssum[tid] = 0.f; ssq[tid] = 0.f; }

  int ntile = blockIdx.x * 8 + w;
  int n0 = ntile * 16;
  int li = lane & 15, hi = lane >> 4;
  int n = n0 + li;
  int b = n / TOUT, t = n - b * TOUT;

  // cooperative A-slice staging: thread stages 16 halves of row (tid>>1)
  const int s_oo = tid >> 1, s_part = tid & 1;
  const _Float16* s_src0 = Abase + (long)s_oo * Kt + s_part * 16;
  _Float16* s_dst0 = &Asl[0][tid * 16];
  _Float16* s_dst1 = &Asl[1][tid * 16];

  auto stage = [&](int ks, int buf) {
    const _Float16* s = s_src0 + ks * 32;
    _Float16* d = buf ? s_dst1 : s_dst0;
#if HAVE_ASYNC_LDS
    async_cp16(s,     d);
    async_cp16(s + 8, d + 8);
#else
    v8h* dv = (v8h*)d;
    dv[0] = *(const v8h*)(s);
    dv[1] = *(const v8h*)(s + 8);
#endif
  };

  v8f acc[8] = {};

  stage(0, 0);
  wait_async0();
  __syncthreads();

  for (int ks = 0; ks < ksteps; ++ks) {
    int cur = ks & 1;
    if (ks + 1 < ksteps) stage(ks + 1, cur ^ 1);

    // B fragment: lane = column (b,t); hi selects K half of the 32-slice.
    int kk = ks*32 + hi*16;
    int j  = kk / DPAD, c0 = kk - j*DPAD;          // never straddles a tap
    int x  = t + j + delta;
    const _Float16* bp = p.capT + ((long)(b*XPAD + x)*DPAD + c0);
    union F { v16h v; v8h h[2]; };
    F B;
    B.h[0] = *(const v8h*)bp;
    B.h[1] = *(const v8h*)(bp + 8);

    // A fragments (contiguous K order, matching B): lane row M=li,
    // elems 0..15 = K[16*hi..+15];
    // software-pipelined so ds_load(mt+1) overlaps wmma(mt).
    const _Float16* ab0 = &Asl[cur][0] + li*32 + 16*hi;
    F Aa, Ab;
    Aa.h[0] = *(const v8h*)(ab0);
    Aa.h[1] = *(const v8h*)(ab0 + 8);
    #pragma unroll
    for (int mt = 0; mt < 8; ++mt) {
      F* curA = (mt & 1) ? &Ab : &Aa;
      F* nxtA = (mt & 1) ? &Aa : &Ab;
      if (mt < 7) {
        const _Float16* ap = ab0 + (mt + 1)*16*32;
        nxtA->h[0] = *(const v8h*)(ap);
        nxtA->h[1] = *(const v8h*)(ap + 8);
      }
      acc[mt] = __builtin_amdgcn_wmma_f32_16x16x32_f16(
          false, curA->v, false, B.v, (short)0, acc[mt], false, false);
    }

    wait_async0();          // staging of ks+1 complete before the barrier
    __syncthreads();        // one barrier per K-step (double-buffered LDS)
  }

  // ---- fused epilogue: per-channel sum/sumsq, per-(b,o) max/min ----------
  const float INF = __int_as_float(0x7f800000);
  int psplit = TOUT - (n0 % TOUT); if (psplit > 16) psplit = 16;  // b boundary
  int b0 = n0 / TOUT;
  long statbase = ((long)img*3 + kidx) * OO;
  long mmbase   = statbase * BC;                   // [o][b] block

  #pragma unroll
  for (int mt = 0; mt < 8; ++mt) {
    #pragma unroll
    for (int r = 0; r < 8; ++r) {
      float v = acc[mt][r];
      int o = mt*16 + r + 8*hi;                    // C/D: VGPR r -> M=r+8*hi
      // sum/sumsq across the 16 columns (xor masks stay inside hi-group)
      float sv = v, qv = v*v;
      for (int m = 1; m < 16; m <<= 1) {
        sv += __shfl_xor(sv, m);
        qv += __shfl_xor(qv, m);
      }
      if (li == 0) { atomicAdd(&ssum[o], sv); atomicAdd(&ssq[o], qv); }
      // segmented max/min over t (columns may span two captions)
      float m0 = (li <  psplit) ? v : -INF;
      float n0v= (li <  psplit) ? v :  INF;
      float m1 = (li >= psplit) ? v : -INF;
      float n1v= (li >= psplit) ? v :  INF;
      for (int m = 1; m < 16; m <<= 1) {
        m0 = fmaxf(m0, __shfl_xor(m0, m)); n0v = fminf(n0v, __shfl_xor(n0v, m));
        m1 = fmaxf(m1, __shfl_xor(m1, m)); n1v = fminf(n1v, __shfl_xor(n1v, m));
      }
      if (li == 0) {
        atomicMaxF(p.gmax + mmbase + (long)o*BC + b0, m0);
        atomicMinF(p.gmin + mmbase + (long)o*BC + b0, n0v);
        if (psplit < 16) {
          atomicMaxF(p.gmax + mmbase + (long)o*BC + b0 + 1, m1);
          atomicMinF(p.gmin + mmbase + (long)o*BC + b0 + 1, n1v);
        }
      }
    }
  }
  __syncthreads();
  if (tid < OO) {
    atomicAdd(p.gsum + statbase + tid, ssum[tid]);
    atomicAdd(p.gsq  + statbase + tid, ssq[tid]);
  }
}

// ---------------------------------------------------------------------------
// K5: feat[img][b][o3] = relu(BN(max/min-selected conv extreme))
// ---------------------------------------------------------------------------
__global__ void __launch_bounds__(256)
k_feat(const float* __restrict__ gsum, const float* __restrict__ gsq,
       const float* __restrict__ gmax, const float* __restrict__ gmin,
       const float* __restrict__ bn1g, const float* __restrict__ bn1b,
       const float* __restrict__ bn2g, const float* __restrict__ bn2b,
       const float* __restrict__ bn3g, const float* __restrict__ bn3b,
       float* __restrict__ feat) {
  long idx = (long)blockIdx.x * 256 + threadIdx.x;   // 32*128*384
  if (idx >= (long)BI*BC*3*OO) return;
  int o3  = (int)(idx % (3*OO));
  long r  = idx / (3*OO);
  int bq  = (int)(r % BC);
  int img = (int)(r / BC);
  int kidx = o3 / OO, o = o3 - kidx*OO;
  long sb = ((long)img*3 + kidx)*OO + o;
  const float inv = 1.f / (float)NCOL;
  float mu  = gsum[sb] * inv;
  float var = gsq[sb] * inv - mu*mu;
  float gg = (kidx==0 ? bn1g : kidx==1 ? bn2g : bn3g)[o];
  float bb = (kidx==0 ? bn1b : kidx==1 ? bn2b : bn3b)[o];
  float sc = gg * rsqrtf(var + EPS);
  float sh = bb - mu * sc;
  float xv = (sc >= 0.f) ? gmax[sb*BC + bq] : gmin[sb*BC + bq];
  feat[idx] = fmaxf(xv * sc + sh, 0.f);
}

// ---------------------------------------------------------------------------
// K6a: h[img][b][f] = feat row @ sim_w1 + b1
// ---------------------------------------------------------------------------
__global__ void __launch_bounds__(256)
k_mlp1(const float* __restrict__ feat, const float* __restrict__ w1,
       const float* __restrict__ b1, float* __restrict__ h) {
  long idx = (long)blockIdx.x * 256 + threadIdx.x;   // 32*128*128
  if (idx >= (long)BI*BC*128) return;
  int f = (int)(idx & 127);
  long r = idx >> 7;
  int bq  = (int)(r % BC);
  int img = (int)(r / BC);
  const float* xr = feat + ((long)img*BC + bq) * (3*OO);
  float a = b1[f];
  for (int j = 0; j < 3*OO; ++j) a += xr[j] * w1[j*128 + f];
  h[idx] = a;
}

// ---------------------------------------------------------------------------
// K6b: per-image BN over batch + ReLU + w2 + sigmoid -> out[img][b]
// ---------------------------------------------------------------------------
__global__ void __launch_bounds__(128)
k_mlp2(const float* __restrict__ h, const float* __restrict__ bng,
       const float* __restrict__ bnb, const float* __restrict__ w2,
       const float* __restrict__ b2, float* __restrict__ out) {
  __shared__ float sc[128], sh[128];
  int img = blockIdx.x, tid = threadIdx.x;
  const float* hp = h + (long)img * BC * 128;
  {
    float s = 0.f, q = 0.f;
    for (int bq = 0; bq < BC; ++bq) { float v = hp[bq*128 + tid]; s += v; q += v*v; }
    float mu = s * (1.f/BC), var = q * (1.f/BC) - mu*mu;
    float scl = bng[tid] * rsqrtf(var + EPS);
    sc[tid] = scl; sh[tid] = bnb[tid] - mu * scl;
  }
  __syncthreads();
  {
    const float* row = hp + tid*128;
    float a = 0.f;
    for (int f = 0; f < 128; ++f) a += fmaxf(row[f]*sc[f] + sh[f], 0.f) * w2[f];
    a += b2[0];
    out[img*BC + tid] = 1.f / (1.f + expf(-a));
  }
}

// ---------------------------------------------------------------------------
extern "C" void kernel_launch(void* const* d_in, const int* in_sizes, int n_in,
                              void* d_out, int out_size, void* d_ws, size_t ws_size,
                              hipStream_t stream) {
  (void)in_sizes; (void)n_in; (void)out_size; (void)ws_size;
  const float* img_embed = (const float*)d_in[0];
  const float* cap_embed = (const float*)d_in[1];
  const float* adapt_w   = (const float*)d_in[2];
  const float* adapt_b   = (const float*)d_in[3];
  const float* k1_w  = (const float*)d_in[4];
  const float* k1_b  = (const float*)d_in[5];
  const float* k1_g  = (const float*)d_in[8];
  const float* k1_be = (const float*)d_in[9];
  const float* k2_w  = (const float*)d_in[10];
  const float* k2_b  = (const float*)d_in[11];
  const float* k2_g  = (const float*)d_in[14];
  const float* k2_be = (const float*)d_in[15];
  const float* k3_w  = (const float*)d_in[16];
  const float* k3_b  = (const float*)d_in[17];
  const float* k3_g  = (const float*)d_in[20];
  const float* k3_be = (const float*)d_in[21];
  const float* bn1_g = (const float*)d_in[22];
  const float* bn1_b = (const float*)d_in[23];
  const float* bn2_g = (const float*)d_in[24];
  const float* bn2_b = (const float*)d_in[25];
  const float* bn3_g = (const float*)d_in[26];
  const float* bn3_b = (const float*)d_in[27];
  const float* sim_w1 = (const float*)d_in[28];
  const float* sim_b1 = (const float*)d_in[29];
  const float* sim_bng= (const float*)d_in[30];
  const float* sim_bnb= (const float*)d_in[31];
  const float* sim_w2 = (const float*)d_in[32];
  const float* sim_b2 = (const float*)d_in[33];
  // d_in[6,7,12,13,18,19] (conv biases) cancel exactly in BN; d_in[34] unused.
  float* out = (float*)d_out;

  // --- workspace carve-up (all offsets 256B aligned) ---
  char* ws = (char*)d_ws;
  size_t off = 0;
  auto alloc = [&](size_t bytes) { void* p = ws + off;
    off += (bytes + 255) & ~(size_t)255; return p; };
  float*    ic    = (float*)   alloc((size_t)BI*HH*4);            // 16KB
  float*    gsum  = (float*)   alloc((size_t)BI*3*OO*4);          // 48KB
  float*    gsq   = (float*)   alloc((size_t)BI*3*OO*4);
  float*    gmax  = (float*)   alloc((size_t)BI*3*OO*BC*4);       // 6.3MB
  float*    gmin  = (float*)   alloc((size_t)BI*3*OO*BC*4);
  float*    feat  = (float*)   alloc((size_t)BI*BC*3*OO*4);       // 6.3MB
  float*    hbuf  = (float*)   alloc((size_t)BI*BC*128*4);        // 2.1MB
  _Float16* kernA = (_Float16*)alloc((size_t)BI*KERN_PER_IMG*2);  // 15.7MB
  _Float16* capT  = (_Float16*)alloc((size_t)BC*XPAD*DPAD*2);     // 5.2MB

  // K0: init accumulators
  {
    long n = (long)BI*3*OO*BC;
    k_init<<<dim3((unsigned)((n + 255)/256)), dim3(256), 0, stream>>>(gsum, gsq, gmax, gmin);
  }
  // K1: adapter
  k_adapt<<<dim3(BI), dim3(256), 0, stream>>>(img_embed, adapt_w, adapt_b, ic);
  // K2: padded/transposed f16 captions
  {
    long n = (long)BC*XPAD*DPAD;
    k_capT<<<dim3((unsigned)((n + 255)/256)), dim3(256), 0, stream>>>(cap_embed, capT);
  }
  // K3: kernel generation + kernel BN
  {
    GenP gp;
    gp.ic = ic;
    gp.kw[0]=k1_w; gp.kw[1]=k2_w; gp.kw[2]=k3_w;
    gp.kb[0]=k1_b; gp.kb[1]=k2_b; gp.kb[2]=k3_b;
    gp.g[0]=k1_g;  gp.g[1]=k2_g;  gp.g[2]=k3_g;
    gp.be[0]=k1_be;gp.be[1]=k2_be;gp.be[2]=k3_be;
    gp.kernA = kernA;
    k_gen<<<dim3(OO, 3, BI), dim3(256), 0, stream>>>(gp);
  }
  // K4: WMMA conv + fused reductions
  {
    ConvP cp; cp.kernA = kernA; cp.capT = capT;
    cp.gsum = gsum; cp.gsq = gsq; cp.gmax = gmax; cp.gmin = gmin;
    k_conv<<<dim3(NTILES/8, 3, BI), dim3(256), 0, stream>>>(cp);
  }
  // K5: BN + ReLU + temporal max -> features
  {
    long n = (long)BI*BC*3*OO;
    k_feat<<<dim3((unsigned)((n + 255)/256)), dim3(256), 0, stream>>>(
        gsum, gsq, gmax, gmin, bn1_g, bn1_b, bn2_g, bn2_b, bn3_g, bn3_b, feat);
  }
  // K6: similarity MLP
  {
    long n = (long)BI*BC*128;
    k_mlp1<<<dim3((unsigned)((n + 255)/256)), dim3(256), 0, stream>>>(feat, sim_w1, sim_b1, hbuf);
    k_mlp2<<<dim3(BI), dim3(128), 0, stream>>>(hbuf, sim_bng, sim_bnb, sim_w2, sim_b2, out);
  }
}
